// DepthEstimator_90520730731178
// MI455X (gfx1250) — matrix operand order repaired and checked by reading, not compiled, not run on hardware
//
#include <hip/hip_runtime.h>

typedef __attribute__((ext_vector_type(16))) __bf16 v16bf;
typedef __attribute__((ext_vector_type(8)))  __bf16 v8bf;
typedef __attribute__((ext_vector_type(8)))  float  v8f;
typedef __attribute__((ext_vector_type(4)))  unsigned int v4u;
typedef __attribute__((ext_vector_type(8)))  int v8i;
typedef __attribute__((ext_vector_type(4)))  int v4i;

#define BDIM 8
#define CINC 256
#define HDIM 96
#define WDIM 160
#define HWp  (HDIM*WDIM)      /* 15360 */
#define NFLAT (BDIM*HWp)      /* 122880 */
#define HP 98                 /* padded height  */
#define WP 168                /* padded width (16B-aligned rows, absorbs over-read) */

// ---------------------------------------------------------------------------
// prep kernels
// ---------------------------------------------------------------------------
__global__ void pad_cvt_kernel(const float* __restrict__ in, __bf16* __restrict__ out, int C) {
    size_t i = (size_t)blockIdx.x * blockDim.x + threadIdx.x;
    size_t total = (size_t)BDIM * C * HP * WP;
    if (i >= total) return;
    int xp = (int)(i % WP);
    size_t t = i / WP;
    int yp = (int)(t % HP);
    size_t bc = t / HP;
    int xx = xp - 1, yy = yp - 1;
    __bf16 v = (__bf16)0.0f;
    if ((unsigned)xx < (unsigned)WDIM && (unsigned)yy < (unsigned)HDIM)
        v = (__bf16)in[(bc * HDIM + yy) * WDIM + xx];
    out[i] = v;
}

__global__ void zero_bf16_kernel(__bf16* __restrict__ p, size_t n) {
    size_t i = (size_t)blockIdx.x * blockDim.x + threadIdx.x;
    if (i < n) p[i] = (__bf16)0.0f;
}

// w: (Co, 256, 3, 3) f32 -> wp: (9, Co, 256) bf16
__global__ void pack_w_kernel(const float* __restrict__ w, __bf16* __restrict__ wp, int Co) {
    int i = blockIdx.x * blockDim.x + threadIdx.x;
    int total = Co * CINC * 9;
    if (i >= total) return;
    int t = i % 9;
    int c = (i / 9) % CINC;
    int o = i / (9 * CINC);
    wp[((size_t)t * Co + o) * CINC + c] = (__bf16)w[i];
}

__global__ void bn_fold_kernel(const float* g, const float* be, const float* m,
                               const float* v, const float* bias, float* a, float* c, int n) {
    int i = blockIdx.x * blockDim.x + threadIdx.x;
    if (i < n) {
        float ai = g[i] * rsqrtf(v[i] + 1e-5f);
        a[i] = ai;
        c[i] = be[i] + ai * (bias[i] - m[i]);
    }
}

__global__ void bias_only_kernel(const float* bias, float* a, float* c, int n) {
    int i = blockIdx.x * blockDim.x + threadIdx.x;
    if (i < n) { a[i] = 1.0f; c[i] = bias[i]; }
}

// ---------------------------------------------------------------------------
// 3x3 conv, implicit GEMM on WMMA bf16 / f32 acc, fused affine + ReLU.
// A (weight) tiles staged with the Tensor Data Mover (tensor_load_to_lds);
// activation slab staged once per 32-ch chunk and shared by all 9 taps.
// ---------------------------------------------------------------------------
template<int COUT, bool PADOUT>
__global__ __launch_bounds__(256, 1)
void conv3x3_wmma(const __bf16* __restrict__ x,
                  const __bf16* __restrict__ wp,
                  const float* __restrict__ abn,
                  const float* __restrict__ cbn,
                  __bf16* __restrict__ y)
{
    __shared__ __attribute__((aligned(16))) __bf16 As[3][128][32]; // [tap][m][k]
    __shared__ __attribute__((aligned(16))) __bf16 Bs[6][40][32];  // [row][col][k]

    const int tid  = threadIdx.x;
    const int lane = tid & 31;
    const int wave = tid >> 5;
    const int wm   = wave & 1;    // M half
    const int wn   = wave >> 1;   // output row within tile (0..3)

    int tb = blockIdx.x;
    const int x0 = (tb % 5) * 32; tb /= 5;
    const int y0 = (tb % 24) * 4; tb /= 24;
    const int b  = tb;            // 0..7
    const int mTile = blockIdx.y * 128;

    const size_t plane = (size_t)HP * WP;
    const __bf16* xbase = x + ((size_t)b * CINC) * plane + (size_t)y0 * WP + x0;

    v8f acc[4][2] = {};

    const int ar   = wm * 64 + (lane & 15);
    const int ak   = (lane >> 4) * 8;
    const int ncol = lane & 15;
    const int bkb  = (lane >> 4) * 8;

    for (int c0 = 0; c0 < CINC; c0 += 32) {
        // ---- stage activation slab: 6 rows x 40 cols x 32 k (cols 34..39 unused pad)
        for (int cidx = tid; cidx < 6 * 32 * 5; cidx += 256) {
            const int cx = cidx % 5;            // 8-col chunk
            const int k  = (cidx / 5) % 32;
            const int ry = cidx / 160;
            const __bf16* src = xbase + (size_t)(c0 + k) * plane + (size_t)ry * WP + cx * 8;
            __builtin_prefetch((const void*)(src + 32 * plane), 0, 0);  // next K chunk
            v8bf vv = *(const v8bf*)src;
            #pragma unroll
            for (int e = 0; e < 8; ++e)
                Bs[ry][cx * 8 + e][k] = vv[e];
        }

        for (int ty = 0; ty < 3; ++ty) {
            // ---- stage A for taps ty*3 .. ty*3+2 via Tensor Data Mover:
            // 2-D tile: 32 (contiguous k, 2B elems) x 128 rows, row stride 256 elems,
            // packed into As[tp] (128x32 bf16, contiguous) in LDS.
            if (wave == 0) {
                #pragma unroll
                for (int tp = 0; tp < 3; ++tp) {
                    const __bf16* gsrc = wp + ((size_t)((ty * 3 + tp) * COUT + mTile)) * CINC + c0;
                    unsigned long long ga = (unsigned long long)(const void*)gsrc;
                    unsigned lds = (unsigned)(size_t)(const void*)&As[tp][0][0];
                    v4u g0;
                    g0[0] = 1u;                                    // count=1, user D#
                    g0[1] = lds;                                   // lds_addr
                    g0[2] = (unsigned)ga;                          // global_addr[31:0]
                    g0[3] = (unsigned)(ga >> 32) | (2u << 30);     // global_addr[56:32] | type=2
                    v8i g1;
                    g1[0] = (int)(1u << 16);                       // data_size = 2 bytes
                    g1[1] = (int)(32u << 16);                      // tensor_dim0 = 32 (lo16 in [31:16])
                    g1[2] = (int)(128u << 16);                     // tensor_dim1 = 128
                    g1[3] = (int)(32u << 16);                      // tile_dim0 = 32
                    g1[4] = 128;                                   // tile_dim1 = 128, tile_dim2 = 0
                    g1[5] = 256;                                   // tensor_dim0_stride = 256 elems
                    g1[6] = 0;
                    g1[7] = 0;
                    v4i zz = {0, 0, 0, 0};
                    v8i z8 = {0, 0, 0, 0, 0, 0, 0, 0};
                    __builtin_amdgcn_tensor_load_to_lds(g0, g1, zz, zz, z8, 0);
                }
                __builtin_amdgcn_s_wait_tensorcnt(0);
            }
            __syncthreads();

            #pragma unroll
            for (int tx = 0; tx < 3; ++tx) {
                // A fragments: 16x32, lane<16 -> M=lane, K halves per ISA striping
                v16bf afr[4];
                #pragma unroll
                for (int i = 0; i < 4; ++i) {
                    const __bf16* p = &As[tx][ar + i * 16][ak];
                    v8bf lo = *(const v8bf*)p;
                    v8bf hi = *(const v8bf*)(p + 16);
                    afr[i] = __builtin_shufflevector(lo, hi,
                             0,1,2,3,4,5,6,7,8,9,10,11,12,13,14,15);
                }
                // B fragments: 32x16, shifted slab view (row wn+ty, col +tx)
                v16bf bfr[2];
                #pragma unroll
                for (int j = 0; j < 2; ++j) {
                    const __bf16* p = &Bs[wn + ty][j * 16 + ncol + tx][bkb];
                    v8bf lo = *(const v8bf*)p;
                    v8bf hi = *(const v8bf*)(p + 16);
                    bfr[j] = __builtin_shufflevector(lo, hi,
                             0,1,2,3,4,5,6,7,8,9,10,11,12,13,14,15);
                }
                #pragma unroll
                for (int i = 0; i < 4; ++i)
                    #pragma unroll
                    for (int j = 0; j < 2; ++j)
                        acc[i][j] = __builtin_amdgcn_wmma_f32_16x16x32_bf16(
                            false, afr[i], false, bfr[j], (short)0, acc[i][j], false, false);
            }
            __syncthreads();
        }
    }

    // ---- epilogue: affine + ReLU, store bf16
    const int mhalf = (lane >> 4) * 8;
    const int oy = y0 + wn;
    #pragma unroll
    for (int i = 0; i < 4; ++i) {
        #pragma unroll
        for (int r = 0; r < 8; ++r) {
            const int m = mTile + wm * 64 + i * 16 + mhalf + r;
            const float a = abn[m];
            const float cc = cbn[m];
            #pragma unroll
            for (int j = 0; j < 2; ++j) {
                const int ox = x0 + j * 16 + ncol;
                float v = a * acc[i][j][r] + cc;
                v = v > 0.0f ? v : 0.0f;
                if (PADOUT)
                    y[(((size_t)(b * COUT + m)) * HP + (oy + 1)) * WP + (ox + 1)] = (__bf16)v;
                else
                    y[(((size_t)(b * COUT + m)) * HDIM + oy) * WDIM + ox] = (__bf16)v;
            }
        }
    }
}

// ---------------------------------------------------------------------------
// fused 1x1 convs + sigmoid->depth + softplus
// ---------------------------------------------------------------------------
__global__ __launch_bounds__(256)
void head_kernel(const __bf16* __restrict__ h2, const __bf16* __restrict__ u1,
                 const float* __restrict__ w3, const float* __restrict__ b3,
                 const float* __restrict__ uw2, const float* __restrict__ ub2,
                 float* __restrict__ depth, float* __restrict__ unc)
{
    __shared__ float sw3[256];
    __shared__ float swu[128];
    int tid = threadIdx.x;
    sw3[tid] = w3[tid];
    if (tid < 128) swu[tid] = uw2[tid];
    __syncthreads();

    int n  = blockIdx.x * 256 + tid;
    int b  = n / HWp;
    int sp = n - b * HWp;

    const __bf16* hp = h2 + (size_t)b * CINC * HWp + sp;
    float s1 = b3[0];
    for (int c = 0; c < 256; ++c) s1 += sw3[c] * (float)hp[(size_t)c * HWp];

    const __bf16* up = u1 + (size_t)b * 128 * HWp + sp;
    float s2 = ub2[0];
    for (int c = 0; c < 128; ++c) s2 += swu[c] * (float)up[(size_t)c * HWp];

    float disp = 1.0f / (1.0f + __expf(-s1));
    const float min_disp = 1.0f / 100.0f;
    const float max_disp = 1.0f / 0.1f;
    depth[n] = 1.0f / (min_disp + (max_disp - min_disp) * disp);
    unc[n]   = (s2 > 20.0f) ? s2 : log1pf(__expf(s2));
}

// ---------------------------------------------------------------------------
// exact lower-median per bbox: binary search on uint bits of positive floats
// ---------------------------------------------------------------------------
__global__ __launch_bounds__(256)
void obj_median_kernel(const float* __restrict__ depth, const int* __restrict__ bb,
                       float* __restrict__ obj)
{
    __shared__ int scnt;
    const int bi = blockIdx.x;
    const int b  = bi >> 6;
    const int* box = bb + bi * 4;
    int x1 = max(box[0], 0), y1 = max(box[1], 0);
    int x2 = min(box[2], WDIM), y2 = min(box[3], HDIM);
    int wbx = x2 - x1, hbx = y2 - y1;
    int area = (wbx > 0 && hbx > 0) ? wbx * hbx : 0;
    if (area == 0) {
        if (threadIdx.x == 0) obj[bi] = 0.0f;
        return;
    }
    const int k = (area - 1) >> 1;
    const float* dimg = depth + (size_t)b * HWp;

    unsigned lo = 0u, hi = 0x7F800000u;
    for (int it = 0; it < 31; ++it) {
        unsigned mid = (lo + hi) >> 1;
        if (threadIdx.x == 0) scnt = 0;
        __syncthreads();
        int part = 0;
        for (int idx = threadIdx.x; idx < area; idx += 256) {
            int yy = y1 + idx / wbx;
            int xx = x1 + idx - (idx / wbx) * wbx;
            unsigned vb = __float_as_uint(dimg[yy * WDIM + xx]);
            part += (vb <= mid) ? 1 : 0;
        }
        atomicAdd(&scnt, part);
        __syncthreads();
        int cle = scnt;
        if (cle >= k + 1) hi = mid; else lo = mid + 1;
        __syncthreads();
    }
    if (threadIdx.x == 0) obj[bi] = __uint_as_float(hi);
}

// ---------------------------------------------------------------------------
// launch
// ---------------------------------------------------------------------------
extern "C" void kernel_launch(void* const* d_in, const int* in_sizes, int n_in,
                              void* d_out, int out_size, void* d_ws, size_t ws_size,
                              hipStream_t stream)
{
    (void)in_sizes; (void)n_in; (void)out_size; (void)ws_size;

    const float* features = (const float*)d_in[0];
    const int*   bboxes   = (const int*)d_in[1];
    const float* w1  = (const float*)d_in[2];
    const float* b1  = (const float*)d_in[3];
    const float* g1  = (const float*)d_in[4];
    const float* be1 = (const float*)d_in[5];
    const float* m1  = (const float*)d_in[6];
    const float* v1  = (const float*)d_in[7];
    const float* w2  = (const float*)d_in[8];
    const float* b2  = (const float*)d_in[9];
    const float* g2  = (const float*)d_in[10];
    const float* be2 = (const float*)d_in[11];
    const float* m2  = (const float*)d_in[12];
    const float* v2  = (const float*)d_in[13];
    const float* w3  = (const float*)d_in[14];
    const float* b3  = (const float*)d_in[15];
    const float* uw1 = (const float*)d_in[16];
    const float* ub1 = (const float*)d_in[17];
    const float* uw2 = (const float*)d_in[18];
    const float* ub2 = (const float*)d_in[19];

    char* ws = (char*)d_ws;
    const size_t padBytes = (size_t)BDIM * CINC * HP * WP * 2;   // 67,436,544
    const size_t u1Bytes  = (size_t)NFLAT * 128 * 2;             // 31,457,280
    __bf16* featp = (__bf16*)(ws);
    __bf16* h1p   = (__bf16*)(ws + padBytes);
    __bf16* u1    = (__bf16*)(ws + 2 * padBytes);
    __bf16* w1p   = (__bf16*)(ws + 2 * padBytes + u1Bytes);
    __bf16* w2p   = (__bf16*)((char*)w1p + (size_t)9 * 256 * CINC * 2);
    __bf16* uw1p  = (__bf16*)((char*)w2p + (size_t)9 * 256 * CINC * 2);
    float*  a1    = (float*)((char*)uw1p + (size_t)9 * 128 * CINC * 2);
    float*  c1    = a1 + 256;
    float*  a2    = c1 + 256;
    float*  c2    = a2 + 256;
    float*  au    = c2 + 256;
    float*  cu    = au + 128;
    __bf16* h2    = featp;   // reuse padded feature buffer once consumed

    float* depth = (float*)d_out;
    float* unc   = depth + NFLAT;
    float* obj   = unc + NFLAT;

    const size_t padElems = (size_t)BDIM * CINC * HP * WP;       // 33,718,272

    // 1. pad + convert features; pre-zero h1p (halo must be 0 for conv2)
    pad_cvt_kernel<<<(unsigned)((padElems + 255) / 256), 256, 0, stream>>>(features, featp, CINC);
    zero_bf16_kernel<<<(unsigned)((padElems + 255) / 256), 256, 0, stream>>>(h1p, padElems);
    // 2. pack weights to (tap, O, C) bf16
    pack_w_kernel<<<(256 * CINC * 9 + 255) / 256, 256, 0, stream>>>(w1, w1p, 256);
    pack_w_kernel<<<(256 * CINC * 9 + 255) / 256, 256, 0, stream>>>(w2, w2p, 256);
    pack_w_kernel<<<(128 * CINC * 9 + 255) / 256, 256, 0, stream>>>(uw1, uw1p, 128);
    // 3. fold BN / bias
    bn_fold_kernel<<<1, 256, 0, stream>>>(g1, be1, m1, v1, b1, a1, c1, 256);
    bn_fold_kernel<<<1, 256, 0, stream>>>(g2, be2, m2, v2, b2, a2, c2, 256);
    bias_only_kernel<<<1, 128, 0, stream>>>(ub1, au, cu, 128);
    // 4. conv1: featp -> h1p (padded out, BN+ReLU)
    conv3x3_wmma<256, true ><<<dim3(960, 2), 256, 0, stream>>>(featp, w1p, a1, c1, h1p);
    // 5. uncertainty conv: featp -> u1 (bias+ReLU) — before featp is reused
    conv3x3_wmma<128, false><<<dim3(960, 1), 256, 0, stream>>>(featp, uw1p, au, cu, u1);
    // 6. conv2: h1p -> h2 (aliases featp)
    conv3x3_wmma<256, false><<<dim3(960, 2), 256, 0, stream>>>(h1p, w2p, a2, c2, h2);
    // 7. heads
    head_kernel<<<NFLAT / 256, 256, 0, stream>>>(h2, u1, w3, b3, uw2, ub2, depth, unc);
    // 8. per-box medians
    obj_median_kernel<<<BDIM * 64, 256, 0, stream>>>(depth, bboxes, obj);
}
